// HAN_42082089566238
// MI455X (gfx1250) — matrix operand order repaired and checked by reading, not compiled, not run on hardware
//
#include <hip/hip_runtime.h>

// HAN on MI455X (gfx1250, wave32).
// GEMMs (GAT projections + semantic attention) run as bf16 WMMA
// (v_wmma_f32_16x16x32_bf16, f32 accumulate) with operands pre-swizzled in
// global memory into WMMA fragment order, so the inner loop is pure
// global_load_b128 + v_wmma. Two independent accumulators break the
// WMMA->WMMA RAW chain so the matrix pipe issues without hazard NOPs.
// Graph softmax/aggregation uses f32 global atomics. MLP head folded to 512x2.

#define NN   20000
#define CC   512
#define HH   8
#define DD   64
#define PP   3
#define EE   320000
#define ETOT (EE + NN)
#define EPSV 1e-5f
#define KCH  (CC / 32)   // 16 K-chunks per 512-wide GEMM

typedef __attribute__((ext_vector_type(16))) __bf16 v16bf;
typedef __attribute__((ext_vector_type(8)))  float  v8f;

union BfFrag { unsigned short u[16]; uint4 q[2]; v16bf v; };

__device__ __forceinline__ unsigned short f2bf(float f) {
  unsigned int u = __float_as_uint(f);
  u += 0x7FFFu + ((u >> 16) & 1u);           // round-to-nearest-even
  return (unsigned short)(u >> 16);
}
__device__ __forceinline__ float wsum(float v) {
#pragma unroll
  for (int o = 16; o > 0; o >>= 1) v += __shfl_xor(v, o, 32);
  return v;
}
// monotone float <-> uint mapping for atomicMax-based segment max
__device__ __forceinline__ unsigned int fenc(float f) {
  unsigned int b = __float_as_uint(f);
  return (b & 0x80000000u) ? ~b : (b | 0x80000000u);
}
__device__ __forceinline__ float fdec(unsigned int u) {
  unsigned int b = (u & 0x80000000u) ? (u & 0x7FFFFFFFu) : ~u;
  return __uint_as_float(b);
}

// ---------------------------------------------------------------- utilities
__global__ void fill_u32_kernel(unsigned int* p, unsigned int v, int n) {
  int i = blockIdx.x * blockDim.x + threadIdx.x;
  if (i < n) p[i] = v;
}

// Pack f32 weight matrices [K=512][N=512] into WMMA B-fragment order:
// dst linear = ((chunk*(N/16) + ng)*32 + lane)*16 + e
// where K = chunk*32 + (lane>>4)*16 + e, n = ng*16 + (lane&15).
__global__ void pack_b_kernel(const float* __restrict__ src,
                              unsigned short* __restrict__ dst, int nmat) {
  int idx = blockIdx.x * blockDim.x + threadIdx.x;
  if (idx >= nmat * CC * CC) return;
  int mat = idx / (CC * CC);
  int r   = idx - mat * CC * CC;
  int e     = r & 15;
  int lane  = (r >> 4) & 31;
  int ng    = (r >> 9) & 31;       // N/16 = 32
  int chunk = r >> 14;             // 16 chunks
  int k = chunk * 32 + (lane >> 4) * 16 + e;
  int n = ng * 16 + (lane & 15);
  dst[idx] = f2bf(src[(size_t)mat * CC * CC + (size_t)k * CC + n]);
}

// ------------------------------------------------------------- layer norm
// one wave per 512-wide row; writes f32 row + bf16 copy in WMMA A-fragment
// order: addr = ((rowgrp*KCH + chunk)*32 + lane')*16 + e,
// lane' = (row&15) + 16*((c>>4)&1), e = c&15 within the chunk.
__global__ __launch_bounds__(256) void ln_kernel(
    const float* __restrict__ x, float* __restrict__ y,
    unsigned short* __restrict__ ybf,
    const float* __restrict__ g, const float* __restrict__ b) {
  int wave = threadIdx.x >> 5, lane = threadIdx.x & 31;
  int row = blockIdx.x * 8 + wave;
  if (row >= NN) return;
  const float* xr = x + (size_t)row * CC;
  float vals[16];
  float s = 0.f;
#pragma unroll
  for (int i = 0; i < 16; ++i) { vals[i] = xr[lane + 32 * i]; s += vals[i]; }
  s = wsum(s);
  float mean = s * (1.f / CC);
  float q = 0.f;
#pragma unroll
  for (int i = 0; i < 16; ++i) { float d = vals[i] - mean; q += d * d; }
  q = wsum(q);
  float inv = rsqrtf(q * (1.f / CC) + EPSV);
  const int rg  = row >> 4, r16 = row & 15;
  const int lp  = r16 + 16 * (lane >> 4);
  const int e   = lane & 15;
#pragma unroll
  for (int i = 0; i < 16; ++i) {
    int c = lane + 32 * i;
    float o = (vals[i] - mean) * inv * g[c] + b[c];
    y[(size_t)row * CC + c] = o;
    ybf[(((size_t)rg * KCH + i) * 32 + lp) * 16 + e] = f2bf(o);
  }
}

// ------------------------------------------------------- bf16 WMMA GEMM
// A: fragment-swizzled [M/16][KCH][32][16], B: fragment-swizzled
// [KCH][N/16][32][16]. C: f32 row-major [M][N]. One wave per 16x16 tile,
// 8 waves/block covering 32(M) x 64(N). Inner loop: 8x b128 load + 2
// independent wmma (dual accumulators -> no WMMA->WMMA RAW hazard).
__global__ __launch_bounds__(256) void gemm_bf16_wmma_kernel(
    const unsigned short* __restrict__ A, const unsigned short* __restrict__ B,
    float* __restrict__ C, int M, int N, int K) {
  const int wave = threadIdx.x >> 5, lane = threadIdx.x & 31;
  const int rg = blockIdx.y * 2 + (wave >> 2);   // row group (16 rows each)
  const int ng = blockIdx.x * 4 + (wave & 3);    // col group (16 cols each)
  const int nchunks = K >> 5;                    // assumed even
  const unsigned short* ap = A + ((size_t)rg * nchunks * 32 + lane) * 16;
  const unsigned short* bp = B + ((size_t)ng * 32 + lane) * 16;
  const size_t astep = 32 * 16;
  const size_t bstep = (size_t)(N >> 4) * 32 * 16;
  v8f acc0 = {0.f, 0.f, 0.f, 0.f, 0.f, 0.f, 0.f, 0.f};
  v8f acc1 = {0.f, 0.f, 0.f, 0.f, 0.f, 0.f, 0.f, 0.f};
#pragma unroll 2
  for (int kc = 0; kc < nchunks; kc += 2) {
    BfFrag fa0, fb0, fa1, fb1;
    fa0.q[0] = ((const uint4*)ap)[0];
    fa0.q[1] = ((const uint4*)ap)[1];
    fb0.q[0] = ((const uint4*)bp)[0];
    fb0.q[1] = ((const uint4*)bp)[1];
    fa1.q[0] = ((const uint4*)(ap + astep))[0];
    fa1.q[1] = ((const uint4*)(ap + astep))[1];
    fb1.q[0] = ((const uint4*)(bp + bstep))[0];
    fb1.q[1] = ((const uint4*)(bp + bstep))[1];
    acc0 = __builtin_amdgcn_wmma_f32_16x16x32_bf16(
        false, fa0.v, false, fb0.v, (short)0, acc0, false, false);
    acc1 = __builtin_amdgcn_wmma_f32_16x16x32_bf16(
        false, fa1.v, false, fb1.v, (short)0, acc1, false, false);
    ap += 2 * astep; bp += 2 * bstep;
  }
  const int hl = lane >> 4, l16 = lane & 15;
  const int n = (ng << 4) + l16;
#pragma unroll
  for (int r = 0; r < 8; ++r) {
    int m = (rg << 4) + hl * 8 + r;
    C[(size_t)m * N + n] = acc0[r] + acc1[r];
  }
}

// -------------------------------------------------------- GAT attention
__global__ void elr_kernel(const float* __restrict__ feat,
                           const float* __restrict__ al,
                           const float* __restrict__ ar,
                           float* __restrict__ el, float* __restrict__ er) {
  int idx = blockIdx.x * blockDim.x + threadIdx.x;
  if (idx >= NN * HH) return;
  int n = idx >> 3, h = idx & 7;
  const float* f  = feat + (size_t)n * CC + h * DD;
  const float* pa = al + h * DD;
  const float* pb = ar + h * DD;
  float sl = 0.f, sr = 0.f;
#pragma unroll 8
  for (int d = 0; d < DD; ++d) { float fv = f[d]; sl += fv * pa[d]; sr += fv * pb[d]; }
  el[idx] = sl; er[idx] = sr;
}

__global__ void edge_logits_kernel(const int* __restrict__ src,
                                   const int* __restrict__ dst,
                                   const float* __restrict__ el,
                                   const float* __restrict__ er,
                                   float* __restrict__ ebuf,
                                   unsigned int* __restrict__ emax) {
  int idx = blockIdx.x * blockDim.x + threadIdx.x;
  if (idx >= ETOT * HH) return;
  int e = idx >> 3, h = idx & 7;
  int s, d;
  if (e < EE) { s = src[e]; d = dst[e]; } else { s = d = e - EE; }
  float v = el[s * HH + h] + er[d * HH + h];
  v = (v > 0.f) ? v : 0.2f * v;             // leaky_relu 0.2
  ebuf[idx] = v;
  atomicMax(&emax[d * HH + h], fenc(v));
}

__global__ void edge_exp_kernel(const int* __restrict__ dst,
                                const unsigned int* __restrict__ emax,
                                float* __restrict__ ebuf,
                                float* __restrict__ denom) {
  int idx = blockIdx.x * blockDim.x + threadIdx.x;
  if (idx >= ETOT * HH) return;
  int e = idx >> 3, h = idx & 7;
  int d = (e < EE) ? dst[e] : (e - EE);
  float ee = __expf(ebuf[idx] - fdec(emax[d * HH + h]));
  ebuf[idx] = ee;
  atomicAdd(&denom[d * HH + h], ee);
}

__global__ void edge_alpha_kernel(const int* __restrict__ dst,
                                  const float* __restrict__ denom,
                                  float* __restrict__ ebuf) {
  int idx = blockIdx.x * blockDim.x + threadIdx.x;
  if (idx >= ETOT * HH) return;
  int e = idx >> 3, h = idx & 7;
  int d = (e < EE) ? dst[e] : (e - EE);
  ebuf[idx] = ebuf[idx] / denom[d * HH + h];
}

// one wave per edge: scatter-add alpha*feat[src] into agg[dst]
__global__ __launch_bounds__(256) void edge_agg_kernel(
    const int* __restrict__ src, const int* __restrict__ dst,
    const float* __restrict__ feat, const float* __restrict__ alpha,
    float* __restrict__ agg) {
  int gw = (blockIdx.x * 256 + threadIdx.x) >> 5;
  int lane = threadIdx.x & 31;
  if (gw >= ETOT) return;
  int s, d;
  if (gw < EE) { s = src[gw]; d = dst[gw]; } else { s = d = gw - EE; }
  const float* fs = feat + (size_t)s * CC;
  float* ad = agg + (size_t)d * CC;
  const float* a8 = alpha + (size_t)gw * HH;
  for (int c = lane; c < CC; c += 32)
    atomicAdd(&ad[c], fs[c] * a8[c >> 6]);
}

__global__ void elu_bias_kernel(const float* __restrict__ agg,
                                const float* __restrict__ gb,
                                float* __restrict__ sem) {
  int idx = blockIdx.x * blockDim.x + threadIdx.x;
  if (idx >= NN * CC) return;
  float v = agg[idx] + gb[idx & (CC - 1)];
  sem[idx] = (v > 0.f) ? v : expm1f(v);
}

// ---------------------------------------------------- semantic attention
__global__ __launch_bounds__(256) void sem_score_kernel(
    const float* __restrict__ t, const float* __restrict__ sb,
    const float* __restrict__ sv, const float* __restrict__ svb,
    float* __restrict__ s, int p) {
  int wave = threadIdx.x >> 5, lane = threadIdx.x & 31;
  int n = blockIdx.x * 8 + wave;
  if (n >= NN) return;
  float acc = 0.f;
  for (int c = lane; c < CC; c += 32)
    acc += tanhf(t[(size_t)n * CC + c] + sb[c]) * sv[c];
  acc = wsum(acc);
  if (lane == 0) s[n * PP + p] = acc + svb[0];
}

__global__ void sem_combine_kernel(const float* __restrict__ s,
                                   const float* __restrict__ z0,
                                   const float* __restrict__ z1,
                                   const float* __restrict__ z2,
                                   float* __restrict__ out) {
  int idx = blockIdx.x * blockDim.x + threadIdx.x;
  if (idx >= NN * CC) return;
  int n = idx >> 9;
  float s0 = s[n * 3], s1 = s[n * 3 + 1], s2 = s[n * 3 + 2];
  float mx = fmaxf(s0, fmaxf(s1, s2));
  float e0 = __expf(s0 - mx), e1 = __expf(s1 - mx), e2 = __expf(s2 - mx);
  float inv = 1.f / (e0 + e1 + e2);
  out[idx] = (e0 * z0[idx] + e1 * z1[idx] + e2 * z2[idx]) * inv;
}

// --------------------------------------------------------- batch norm
__global__ __launch_bounds__(512) void bn_stats_kernel(
    const float* __restrict__ x, float* __restrict__ cs, float* __restrict__ cq) {
  int c = threadIdx.x;                 // column 0..511
  int n0 = blockIdx.x * 100;           // 200 blocks * 100 rows = 20000
  float s = 0.f, q = 0.f;
  for (int r = 0; r < 100; ++r) {
    float v = x[(size_t)(n0 + r) * CC + c];
    s += v; q += v * v;
  }
  atomicAdd(&cs[c], s);
  atomicAdd(&cq[c], q);
}
__global__ void bn_apply_kernel(const float* __restrict__ x,
                                const float* __restrict__ cs,
                                const float* __restrict__ cq,
                                const float* __restrict__ g,
                                const float* __restrict__ b,
                                float* __restrict__ y) {
  int idx = blockIdx.x * blockDim.x + threadIdx.x;
  if (idx >= NN * CC) return;
  int c = idx & (CC - 1);
  float m = cs[c] * (1.f / NN);
  float var = cq[c] * (1.f / NN) - m * m;
  y[idx] = (x[idx] - m) * rsqrtf(var + EPSV) * g[c] + b[c];
}

// --------------------------------------------------------- MLP head (folded)
__global__ void mlp_fold_kernel(const float* __restrict__ w1,
                                const float* __restrict__ b1,
                                const float* __restrict__ w2,
                                const float* __restrict__ b2,
                                float* __restrict__ w12, float* __restrict__ b12) {
  int idx = blockIdx.x * blockDim.x + threadIdx.x;
  if (idx < CC * 2) {
    int k = idx >> 1, o = idx & 1;
    float acc = 0.f;
    for (int j = 0; j < 1024; ++j) acc += w1[(size_t)k * 1024 + j] * w2[j * 2 + o];
    w12[idx] = acc;
  } else if (idx < CC * 2 + 2) {
    int o = idx - CC * 2;
    float acc = b2[o];
    for (int j = 0; j < 1024; ++j) acc += b1[j] * w2[j * 2 + o];
    b12[o] = acc;
  }
}
__global__ __launch_bounds__(256) void logits_kernel(
    const float* __restrict__ h, const float* __restrict__ w12,
    const float* __restrict__ b12, float* __restrict__ out) {
  int wave = threadIdx.x >> 5, lane = threadIdx.x & 31;
  int n = blockIdx.x * 8 + wave;
  if (n >= NN) return;
  float a0 = 0.f, a1 = 0.f;
  for (int c = lane; c < CC; c += 32) {
    float hv = h[(size_t)n * CC + c];
    a0 += hv * w12[c * 2];
    a1 += hv * w12[c * 2 + 1];
  }
  a0 = wsum(a0); a1 = wsum(a1);
  if (lane == 0) { out[n * 2] = a0 + b12[0]; out[n * 2 + 1] = a1 + b12[1]; }
}

// ================================================================ launch
extern "C" void kernel_launch(void* const* d_in, const int* in_sizes, int n_in,
                              void* d_out, int out_size, void* d_ws, size_t ws_size,
                              hipStream_t stream) {
  (void)in_sizes; (void)n_in; (void)out_size; (void)ws_size;
  const float* h_in = (const float*)d_in[0];
  const int*   eidx = (const int*)d_in[1];
  const float* ln_g = (const float*)d_in[2];
  const float* ln_b = (const float*)d_in[3];
  const float* Wg   = (const float*)d_in[4];
  const float* al   = (const float*)d_in[5];
  const float* ar   = (const float*)d_in[6];
  const float* gb   = (const float*)d_in[7];
  const float* sW   = (const float*)d_in[8];
  const float* sb   = (const float*)d_in[9];
  const float* sv   = (const float*)d_in[10];
  const float* svb  = (const float*)d_in[11];
  const float* bn_g = (const float*)d_in[12];
  const float* bn_b = (const float*)d_in[13];
  const float* w1   = (const float*)d_in[14];
  const float* b1   = (const float*)d_in[15];
  const float* w2   = (const float*)d_in[16];
  const float* b2   = (const float*)d_in[17];

  // workspace carve-up (~284 MB)
  char* wp = (char*)d_ws;
  auto carve = [&](size_t bytes) {
    void* p = (void*)wp;
    wp += (bytes + 255) & ~(size_t)255;
    return p;
  };
  const size_t NC = (size_t)NN * CC;
  float*          hbuf   = (float*)carve(NC * 4);
  float*          feat   = (float*)carve(NC * 4);   // GAT feat / combined z / BN input
  float*          agg    = (float*)carve(NC * 4);   // GAT agg / semantic GEMM tmp
  float*          sem0   = (float*)carve(NC * 4);
  float*          sem1   = (float*)carve(NC * 4);
  float*          sem2   = (float*)carve(NC * 4);
  unsigned short* hbf    = (unsigned short*)carve(NC * 2);  // A-fragment swizzled
  float*          ebuf   = (float*)carve((size_t)ETOT * HH * 4);
  float*          el     = (float*)carve((size_t)NN * HH * 4);
  float*          er     = (float*)carve((size_t)NN * HH * 4);
  unsigned int*   emax   = (unsigned int*)carve((size_t)NN * HH * 4);
  float*          denom  = (float*)carve((size_t)NN * HH * 4);
  float*          sbuf   = (float*)carve((size_t)NN * PP * 4);
  unsigned short* WgB    = (unsigned short*)carve((size_t)2 * PP * CC * CC * 2);
  unsigned short* sWB    = (unsigned short*)carve((size_t)2 * CC * CC * 2);
  float*          w12    = (float*)carve(CC * 2 * 4);
  float*          b12    = (float*)carve(2 * 4);
  float*          colsum = (float*)carve(CC * 4);
  float*          colsq  = (float*)carve(CC * 4);

  // weight preprocessing: bf16 fragment-swizzled copies + folded MLP
  {
    int nWg = 2 * PP * CC * CC;                       // 1,572,864
    pack_b_kernel<<<(nWg + 255) / 256, 256, 0, stream>>>(Wg, WgB, 2 * PP);
    int nsW = 2 * CC * CC;                            // 524,288
    pack_b_kernel<<<(nsW + 255) / 256, 256, 0, stream>>>(sW, sWB, 2);
    mlp_fold_kernel<<<5, 256, 0, stream>>>(w1, b1, w2, b2, w12, b12);
  }

  const int EH = ETOT * HH;                           // 2,720,000
  float* semP[3] = {sem0, sem1, sem2};

  for (int l = 0; l < 2; ++l) {
    const float* lw_g = ln_g + l * CC;
    const float* lw_b = ln_b + l * CC;
    for (int p = 0; p < PP; ++p) {
      const int* srcp = eidx + (size_t)(p * 2 + 0) * EE;
      const int* dstp = eidx + (size_t)(p * 2 + 1) * EE;
      const float* ln_in = (l == 0 && p == 0) ? h_in : hbuf;
      ln_kernel<<<NN / 8, 256, 0, stream>>>(ln_in, hbuf, hbf, lw_g, lw_b);
      gemm_bf16_wmma_kernel<<<dim3(CC / 64, NN / 32), 256, 0, stream>>>(
          hbf, WgB + (size_t)(l * PP + p) * CC * CC, feat, NN, CC, CC);
      elr_kernel<<<(NN * HH) / 256, 256, 0, stream>>>(
          feat, al + (size_t)(l * PP + p) * CC, ar + (size_t)(l * PP + p) * CC, el, er);
      fill_u32_kernel<<<(NN * HH + 255) / 256, 256, 0, stream>>>(emax, 0u, NN * HH);
      fill_u32_kernel<<<(NN * HH + 255) / 256, 256, 0, stream>>>((unsigned int*)denom, 0u, NN * HH);
      fill_u32_kernel<<<(int)(NC + 255) / 256, 256, 0, stream>>>((unsigned int*)agg, 0u, (int)NC);
      edge_logits_kernel<<<(EH + 255) / 256, 256, 0, stream>>>(srcp, dstp, el, er, ebuf, emax);
      edge_exp_kernel<<<(EH + 255) / 256, 256, 0, stream>>>(dstp, emax, ebuf, denom);
      edge_alpha_kernel<<<(EH + 255) / 256, 256, 0, stream>>>(dstp, denom, ebuf);
      edge_agg_kernel<<<(ETOT * 32 + 255) / 256, 256, 0, stream>>>(srcp, dstp, feat, ebuf, agg);
      elu_bias_kernel<<<(int)(NC + 255) / 256, 256, 0, stream>>>(
          agg, gb + (size_t)(l * PP + p) * CC, semP[p]);
    }
    // semantic attention: z = LN(sem_p); t = tanh(z @ sW + sb); s = t @ sv + svb
    for (int p = 0; p < PP; ++p) {
      ln_kernel<<<NN / 8, 256, 0, stream>>>(semP[p], semP[p], hbf, lw_g, lw_b);
      gemm_bf16_wmma_kernel<<<dim3(CC / 64, NN / 32), 256, 0, stream>>>(
          hbf, sWB + (size_t)l * CC * CC, agg, NN, CC, CC);
      sem_score_kernel<<<NN / 8, 256, 0, stream>>>(agg, sb + l * CC, sv + l * CC, svb + l, sbuf, p);
    }
    sem_combine_kernel<<<(int)(NC + 255) / 256, 256, 0, stream>>>(sbuf, sem0, sem1, sem2, feat);
    // shared BatchNorm over N
    fill_u32_kernel<<<2, 256, 0, stream>>>((unsigned int*)colsum, 0u, CC);
    fill_u32_kernel<<<2, 256, 0, stream>>>((unsigned int*)colsq, 0u, CC);
    bn_stats_kernel<<<NN / 100, 512, 0, stream>>>(feat, colsum, colsq);
    bn_apply_kernel<<<(int)(NC + 255) / 256, 256, 0, stream>>>(feat, colsum, colsq, bn_g, bn_b, hbuf);
  }

  // outputs: h (N*512) then logits (N*2)
  hipMemcpyAsync(d_out, hbuf, NC * 4, hipMemcpyDeviceToDevice, stream);
  logits_kernel<<<NN / 8, 256, 0, stream>>>(hbuf, w12, b12, (float*)d_out + NC);
}
